// MultiHeadAttention_65893388255247
// MI455X (gfx1250) — matrix operand order repaired and checked by reading, not compiled, not run on hardware
//
#include <hip/hip_runtime.h>

// ---------------------------------------------------------------------------
// MHA forward for MI455X (gfx1250, wave32, WMMA bf16 16x16x32 + TDM tile loads).
// B=2, S=2048, E=2048, H=16, HD=128.
// ---------------------------------------------------------------------------

#define Bc  2
#define Sc  2048
#define Ec  2048
#define Hc  16
#define HDc 128
#define Mc  (Bc * Sc)   // 4096

typedef __attribute__((ext_vector_type(16))) __bf16 v16bf;
typedef __attribute__((ext_vector_type(8)))  float  v8f;
typedef __attribute__((ext_vector_type(4)))  unsigned v4u;
typedef __attribute__((ext_vector_type(8)))  int    v8i;
typedef __attribute__((ext_vector_type(4)))  int    v4i;

union FragBF { v16bf v; uint4 q[2]; };
union AccU   { v8f v; float f[8]; };

#if __has_builtin(__builtin_amdgcn_tensor_load_to_lds)
#define HAVE_TDM 1
#else
#define HAVE_TDM 0
#endif

__device__ __forceinline__ unsigned short f2bf_bits(float f) {
  union { float f; unsigned u; } x; x.f = f;
  unsigned r = x.u + 0x7FFFu + ((x.u >> 16) & 1u);   // round-to-nearest-even
  return (unsigned short)(r >> 16);
}
__device__ __forceinline__ __bf16 f2bf(float f) {
  union { unsigned short s; __bf16 b; } u; u.s = f2bf_bits(f); return u.b;
}

// A-matrix fragment (16x32 bf16): lane<16 holds K[0:8)+K[16:24), lane>=16 K[8:16)+K[24:32)
__device__ __forceinline__ v16bf load_fragA(const __bf16* row, int lane) {
  FragBF u;
  const int k0 = (lane & 16) ? 8 : 0;
  u.q[0] = *(const uint4*)(row + k0);
  u.q[1] = *(const uint4*)(row + k0 + 16);
  return u.v;
}
// B-matrix fragment (32x16 bf16): lane is column n; lane<16 holds K[0:8)+K[16:24), lane>=16 K[8:16)+K[24:32)
__device__ __forceinline__ v16bf load_fragB(const __bf16* row, int lane) {
  FragBF u;
  const int k0 = (lane & 16) ? 8 : 0;
  u.q[0] = *(const uint4*)(row + k0);
  u.q[1] = *(const uint4*)(row + k0 + 16);
  return u.v;
}

__device__ __forceinline__ v8f wmma_bf16(v16bf a, v16bf b, v8f c) {
  return __builtin_amdgcn_wmma_f32_16x16x32_bf16(false, a, false, b, (short)0, c,
                                                 false, false);
}

// convert 16 floats -> 16 bf16, store as two b128 to LDS
__device__ __forceinline__ void cvt_store16(__bf16* dst, const float* t) {
  union { uint4 q; unsigned u[4]; } p0, p1;
#pragma unroll
  for (int e = 0; e < 4; e++) {
    p0.u[e] = (unsigned)f2bf_bits(t[2 * e])     | ((unsigned)f2bf_bits(t[2 * e + 1]) << 16);
    p1.u[e] = (unsigned)f2bf_bits(t[8 + 2 * e]) | ((unsigned)f2bf_bits(t[8 + 2 * e + 1]) << 16);
  }
  *(uint4*)dst       = p0.q;
  *(uint4*)(dst + 8) = p1.q;
}

#if HAVE_TDM
// Issue a 2D TDM tile load (bf16 elements) global -> LDS.  Per-wave instruction,
// tracked by TENSORcnt; descriptor packed per CDNA5 ISA.
__device__ __forceinline__ void tdm_load_2d_bf16(
    const void* gaddr, unsigned lds_off,
    unsigned tensor_d0, unsigned tensor_d1, unsigned stride_d0,
    unsigned tile_d0, unsigned tile_d1)
{
  const unsigned long long ga = (unsigned long long)(size_t)gaddr;
  v4u g0;
  g0.x = 1u;                                                // count=1 (valid)
  g0.y = lds_off;                                           // lds_addr (bytes)
  g0.z = (unsigned)(ga & 0xFFFFFFFFu);                      // global_addr[31:0]
  g0.w = (unsigned)((ga >> 32) & 0x01FFFFFFu) | 0x80000000u;// addr[56:32] | type=2
  v8i g1;
  g1[0] = (int)(1u << 16);                                  // data_size=1 (2 bytes)
  g1[1] = (int)((tensor_d0 & 0xFFFFu) << 16);               // tensor_dim0 lo
  g1[2] = (int)((tensor_d0 >> 16) | ((tensor_d1 & 0xFFFFu) << 16));
  g1[3] = (int)((tensor_d1 >> 16) | ((tile_d0 & 0xFFFFu) << 16));
  g1[4] = (int)(tile_d1 & 0xFFFFu);                         // tile_dim1, tile_dim2=0
  g1[5] = (int)stride_d0;                                   // tensor_dim0_stride lo
  g1[6] = 0; g1[7] = 0;
  v4i z4 = (v4i)0;
#if defined(__clang_major__) && (__clang_major__ >= 23)
  v8i z8 = (v8i)0;
  __builtin_amdgcn_tensor_load_to_lds(g0, g1, z4, z4, z8, 0);
#else
  __builtin_amdgcn_tensor_load_to_lds(g0, g1, z4, z4, 0);
#endif
}
#endif

// ---------------------------------------------------------------------------
// Kernel 1: Q/K/V = x @ W^T + b, RoPE applied to Q,K (Q also scaled 1/sqrt(HD)),
// bf16 results stored in [B, H, S, HD] layout.  grid = (M/128, E/128, 3)
// ---------------------------------------------------------------------------
__global__ __launch_bounds__(256) void qkv_rope_kernel(
    const float* __restrict__ x,
    const float* __restrict__ wq, const float* __restrict__ bq,
    const float* __restrict__ wk, const float* __restrict__ bk,
    const float* __restrict__ wv, const float* __restrict__ bv,
    const int*   __restrict__ pos_ids,
    __bf16* __restrict__ Qw, __bf16* __restrict__ Kw, __bf16* __restrict__ Vw)
{
  __shared__ __align__(16) __bf16 sA[128 * 32];
  __shared__ __align__(16) __bf16 sB[128 * 32];

  const int tid  = threadIdx.x;
  const int lane = tid & 31;
  const int wave = tid >> 5;
  const int mBase = blockIdx.x * 128;
  const int nBase = blockIdx.y * 128;
  const int mat   = blockIdx.z;                    // 0=Q 1=K 2=V

  const float* W  = (mat == 0) ? wq : (mat == 1) ? wk : wv;
  const float* Bs = (mat == 0) ? bq : (mat == 1) ? bk : bv;
  __bf16*      Out = (mat == 0) ? Qw : (mat == 1) ? Kw : Vw;

  const int m0 = (wave >> 1) * 32;                 // 4 waves over M
  const int n0 = (wave & 1) * 64;                  // 2 waves over N

  v8f acc[2][4];
#pragma unroll
  for (int i = 0; i < 2; i++)
#pragma unroll
    for (int j = 0; j < 4; j++) acc[i][j] = (v8f)(0.0f);

  const int ldRow  = tid >> 1;                     // 0..127
  const int ldHalf = (tid & 1) * 16;               // 0 / 16

  for (int kt = 0; kt < Ec; kt += 32) {
    {
      const float4* s4 = (const float4*)(x + (size_t)(mBase + ldRow) * Ec + kt + ldHalf);
      float t[16];
      *(float4*)(t + 0) = s4[0]; *(float4*)(t + 4)  = s4[1];
      *(float4*)(t + 8) = s4[2]; *(float4*)(t + 12) = s4[3];
      cvt_store16(sA + ldRow * 32 + ldHalf, t);
    }
    {
      const float4* s4 = (const float4*)(W + (size_t)(nBase + ldRow) * Ec + kt + ldHalf);
      float t[16];
      *(float4*)(t + 0) = s4[0]; *(float4*)(t + 4)  = s4[1];
      *(float4*)(t + 8) = s4[2]; *(float4*)(t + 12) = s4[3];
      cvt_store16(sB + ldRow * 32 + ldHalf, t);
    }
    __syncthreads();

    v16bf aF[2], bF[4];
#pragma unroll
    for (int i = 0; i < 2; i++)
      aF[i] = load_fragA(sA + (m0 + i * 16 + (lane & 15)) * 32, lane);
#pragma unroll
    for (int j = 0; j < 4; j++)
      bF[j] = load_fragB(sB + (n0 + j * 16 + (lane & 15)) * 32, lane);
#pragma unroll
    for (int i = 0; i < 2; i++)
#pragma unroll
      for (int j = 0; j < 4; j++) acc[i][j] = wmma_bf16(aF[i], bF[j], acc[i][j]);

    __syncthreads();
  }

  // epilogue: bias (+ RoPE for Q/K) + bf16 store to [B,H,S,HD]
  const int col0  = lane & 15;
  const int rhalf = (lane & 16) ? 8 : 0;
#pragma unroll
  for (int j = 0; j < 4; j++) {
    const int   n    = nBase + n0 + j * 16 + col0;
    const float bias = Bs[n];
    const int   hh   = n >> 7;
    const int   d    = n & 127;
    const float sgn  = (n & 1) ? 1.0f : -1.0f;
    float invf = 0.0f;
    if (mat < 2)  // inv_freq = 10000^(-(n&~1)/HD)
      invf = __expf(-(float)(n & ~1) * (9.210340371976184f / 128.0f));
#pragma unroll
    for (int i = 0; i < 2; i++) {
      AccU u; u.v = acc[i][j];
#pragma unroll
      for (int r = 0; r < 8; r++) {
        const int m = mBase + m0 + i * 16 + r + rhalf;
        float val = u.f[r] + bias;
        if (mat < 2) {
          const float partner = __shfl_xor(val, 1);   // rotate-half pair lives next lane
          const float ang = (float)pos_ids[m] * invf;
          val = val * __cosf(ang) + sgn * partner * __sinf(ang);
          if (mat == 0) val *= 0.08838834764831845f;  // 1/sqrt(HD)
        }
        const int b = m >> 11, s = m & 2047;
        Out[(((size_t)(b * Hc + hh)) * Sc + s) * HDc + d] = f2bf(val);
      }
    }
  }
}

// ---------------------------------------------------------------------------
// Kernel 2: causal flash attention per (b,h), 128-query tiles, 8 waves x 16 rows.
// K and V tiles are DMA'd into LDS by the Tensor Data Mover (TENSORcnt).
// grid = (S/128, H, B)
// ---------------------------------------------------------------------------
__global__ __launch_bounds__(256) void attn_kernel(
    const __bf16* __restrict__ Qw, const __bf16* __restrict__ Kw,
    const __bf16* __restrict__ Vw, const int* __restrict__ amask,
    __bf16* __restrict__ Ab)
{
  __shared__ __align__(16) __bf16 sK[128 * 128];   // K tile row-major [kv][d]
  __shared__ __align__(16) __bf16 sVP[128 * 128];  // V row-major, later reused for P
  __shared__ __align__(16) __bf16 sVT[128 * 128];  // V transposed [d][kv]
  __shared__ int sAM[128];

  const int tid = threadIdx.x, lane = tid & 31, wave = tid >> 5;
  const int qTile = blockIdx.x, hh = blockIdx.y, b = blockIdx.z;
  const int qBase = qTile * 128;
  const size_t headOff = ((size_t)(b * Hc + hh)) * Sc * HDc;
  const __bf16* Qh = Qw + headOff;
  const __bf16* Kh = Kw + headOff;
  const __bf16* Vh = Vw + headOff;

  const int col0  = lane & 15;
  const int rhalf = (lane & 16) ? 8 : 0;
  const int qRow0 = qBase + wave * 16;

  // Q A-fragments for the wave's 16 rows, whole HD, kept in registers
  v16bf qF[4];
  {
    const __bf16* qp = Qh + (size_t)(qRow0 + col0) * HDc;
#pragma unroll
    for (int kk = 0; kk < 4; kk++) qF[kk] = load_fragA(qp + kk * 32, lane);
  }

  v8f oacc[8];
  float mrow[8], lrow[8];
#pragma unroll
  for (int f = 0; f < 8; f++) oacc[f] = (v8f)(0.0f);
#pragma unroll
  for (int r = 0; r < 8; r++) { mrow[r] = -1e30f; lrow[r] = 0.0f; }

  const int nKV = qTile + 1;                       // causal: tiles 0..qTile
  for (int t = 0; t < nKV; t++) {
    const int kvBase = t * 128;
    __syncthreads();                               // prior tile fully consumed

    // ---- stage K,V tiles into LDS via the Tensor Data Mover ----
#if HAVE_TDM
    if (wave == 0) {
      tdm_load_2d_bf16(Kh + (size_t)kvBase * HDc, (unsigned)(size_t)(void*)sK,
                       HDc, Sc, HDc, HDc, 128);
      tdm_load_2d_bf16(Vh + (size_t)kvBase * HDc, (unsigned)(size_t)(void*)sVP,
                       HDc, Sc, HDc, HDc, 128);
      __builtin_amdgcn_s_wait_tensorcnt((short)0);
    }
#else
    for (int idx = tid; idx < 128 * 16; idx += 256) {
      const int r  = idx >> 4;
      const int c8 = (idx & 15) * 8;
      *(uint4*)(sK  + r * HDc + c8) = *(const uint4*)(Kh + (size_t)(kvBase + r) * HDc + c8);
      *(uint4*)(sVP + r * HDc + c8) = *(const uint4*)(Vh + (size_t)(kvBase + r) * HDc + c8);
    }
#endif
    __syncthreads();                               // TDM data visible to all waves

    // ---- transpose V (LDS->LDS), key mask, prefetch next K tile ----
    for (int idx = tid; idx < 128 * 16; idx += 256) {
      const int r  = idx >> 4;                     // kv row
      const int c8 = (idx & 15) * 8;               // d start
      uint4 dv = *(const uint4*)(sVP + r * HDc + c8);
      const __bf16* e = (const __bf16*)&dv;
#pragma unroll
      for (int q = 0; q < 8; q++) sVT[(c8 + q) * 128 + r] = e[q];
      if (t + 1 < nKV)                             // gfx1250 global_prefetch_b8
        __builtin_prefetch(Kh + (size_t)(kvBase + 128 + r) * HDc + c8, 0, 1);
    }
    for (int i = tid; i < 128; i += 256) sAM[i] = amask[b * Sc + kvBase + i];
    __syncthreads();

    // ---- scores = Q * K^T (K B-fragments from LDS: contiguous d) ----
    v8f sc[8];
#pragma unroll
    for (int f = 0; f < 8; f++) sc[f] = (v8f)(0.0f);
#pragma unroll
    for (int f = 0; f < 8; f++) {
      const __bf16* kp = sK + (size_t)(f * 16 + col0) * HDc;
#pragma unroll
      for (int kk = 0; kk < 4; kk++)
        sc[f] = wmma_bf16(qF[kk], load_fragB(kp + kk * 32, lane), sc[f]);
    }

    // ---- causal + key-padding mask, tile row max ----
    float tmax[8];
#pragma unroll
    for (int r = 0; r < 8; r++) tmax[r] = -1e30f;
    AccU su[8];
#pragma unroll
    for (int f = 0; f < 8; f++) {
      su[f].v = sc[f];
      const int  kv    = kvBase + f * 16 + col0;
      const bool keyOk = (sAM[f * 16 + col0] != 0);
#pragma unroll
      for (int r = 0; r < 8; r++) {
        const int q = qRow0 + r + rhalf;
        float v = su[f].f[r];
        if (!keyOk || kv > q) v = -1e30f;
        su[f].f[r] = v;
        tmax[r] = fmaxf(tmax[r], v);
      }
    }
#pragma unroll
    for (int r = 0; r < 8; r++) {
      float v = tmax[r];
      v = fmaxf(v, __shfl_xor(v, 1)); v = fmaxf(v, __shfl_xor(v, 2));
      v = fmaxf(v, __shfl_xor(v, 4)); v = fmaxf(v, __shfl_xor(v, 8));
      tmax[r] = v;                                 // per-half == per-row
    }
    float scale[8], tsum[8];
#pragma unroll
    for (int r = 0; r < 8; r++) {
      const float mn = fmaxf(mrow[r], tmax[r]);
      scale[r] = __expf(mrow[r] - mn);
      mrow[r]  = mn;
      tsum[r]  = 0.0f;
    }
#pragma unroll
    for (int f = 0; f < 8; f++)
#pragma unroll
      for (int r = 0; r < 8; r++) {
        const float sv = su[f].f[r];
        const float p  = (sv <= -1e29f) ? 0.0f : __expf(sv - mrow[r]);
        su[f].f[r] = p;
        tsum[r] += p;
      }
#pragma unroll
    for (int r = 0; r < 8; r++) {
      float v = tsum[r];
      v += __shfl_xor(v, 1); v += __shfl_xor(v, 2);
      v += __shfl_xor(v, 4); v += __shfl_xor(v, 8);
      lrow[r] = lrow[r] * scale[r] + v;
    }
    // rescale running output
#pragma unroll
    for (int f = 0; f < 8; f++) {
      AccU u; u.v = oacc[f];
#pragma unroll
      for (int r = 0; r < 8; r++) u.f[r] *= scale[r];
      oacc[f] = u.v;
    }
    // P -> LDS (C-layout to A-layout reshape; wave-private rows; sVP is dead as V now)
#pragma unroll
    for (int f = 0; f < 8; f++)
#pragma unroll
      for (int r = 0; r < 8; r++)
        sVP[(wave * 16 + r + rhalf) * 128 + f * 16 + col0] = f2bf(su[f].f[r]);

    // ---- O += P * V (B-fragments from transposed V tile: contiguous kv) ----
#pragma unroll
    for (int kk = 0; kk < 4; kk++) {
      const v16bf aP = load_fragA(sVP + (wave * 16 + col0) * 128 + kk * 32, lane);
#pragma unroll
      for (int f = 0; f < 8; f++)
        oacc[f] = wmma_bf16(aP, load_fragB(sVT + (f * 16 + col0) * 128 + kk * 32, lane),
                            oacc[f]);
    }
  }

  // finalize: /l, zero padded queries, store bf16 to [B,S,H,HD] (== [M,E])
#pragma unroll
  for (int f = 0; f < 8; f++) {
    AccU u; u.v = oacc[f];
#pragma unroll
    for (int r = 0; r < 8; r++) {
      const int q   = qRow0 + r + rhalf;
      const int qOk = amask[b * Sc + q];
      const float inv = (lrow[r] > 0.0f) ? 1.0f / lrow[r] : 0.0f;
      const float v = qOk ? u.f[r] * inv : 0.0f;
      Ab[(((size_t)b * Sc + q) * Hc + hh) * HDc + f * 16 + col0] = f2bf(v);
    }
  }
}

// ---------------------------------------------------------------------------
// Kernel 3: out = attn @ wo^T + bo  (A is bf16 workspace, output fp32)
// grid = (M/128, E/128)
// ---------------------------------------------------------------------------
__global__ __launch_bounds__(256) void oproj_kernel(
    const __bf16* __restrict__ A, const float* __restrict__ wo,
    const float* __restrict__ bo, float* __restrict__ out)
{
  __shared__ __align__(16) __bf16 sA[128 * 32];
  __shared__ __align__(16) __bf16 sB[128 * 32];

  const int tid = threadIdx.x, lane = tid & 31, wave = tid >> 5;
  const int mBase = blockIdx.x * 128;
  const int nBase = blockIdx.y * 128;
  const int m0 = (wave >> 1) * 32;
  const int n0 = (wave & 1) * 64;

  v8f acc[2][4];
#pragma unroll
  for (int i = 0; i < 2; i++)
#pragma unroll
    for (int j = 0; j < 4; j++) acc[i][j] = (v8f)(0.0f);

  const int ldRow  = tid >> 1;
  const int ldHalf = (tid & 1) * 16;

  for (int kt = 0; kt < Ec; kt += 32) {
    {
      const __bf16* src = A + (size_t)(mBase + ldRow) * Ec + kt + ldHalf;
      __bf16* dst = sA + ldRow * 32 + ldHalf;
      *(uint4*)dst       = *(const uint4*)src;
      *(uint4*)(dst + 8) = *(const uint4*)(src + 8);
    }
    {
      const float4* s4 = (const float4*)(wo + (size_t)(nBase + ldRow) * Ec + kt + ldHalf);
      float t[16];
      *(float4*)(t + 0) = s4[0]; *(float4*)(t + 4)  = s4[1];
      *(float4*)(t + 8) = s4[2]; *(float4*)(t + 12) = s4[3];
      cvt_store16(sB + ldRow * 32 + ldHalf, t);
    }
    __syncthreads();

    v16bf aF[2], bF[4];
#pragma unroll
    for (int i = 0; i < 2; i++)
      aF[i] = load_fragA(sA + (m0 + i * 16 + (lane & 15)) * 32, lane);
#pragma unroll
    for (int j = 0; j < 4; j++)
      bF[j] = load_fragB(sB + (n0 + j * 16 + (lane & 15)) * 32, lane);
#pragma unroll
    for (int i = 0; i < 2; i++)
#pragma unroll
      for (int j = 0; j < 4; j++) acc[i][j] = wmma_bf16(aF[i], bF[j], acc[i][j]);

    __syncthreads();
  }

  const int col0  = lane & 15;
  const int rhalf = (lane & 16) ? 8 : 0;
#pragma unroll
  for (int j = 0; j < 4; j++) {
    const int   n    = nBase + n0 + j * 16 + col0;
    const float bias = bo[n];
#pragma unroll
    for (int i = 0; i < 2; i++) {
      AccU u; u.v = acc[i][j];
#pragma unroll
      for (int r = 0; r < 8; r++) {
        const int m = mBase + m0 + i * 16 + r + rhalf;
        out[(size_t)m * Ec + n] = u.f[r] + bias;
      }
    }
  }
}

// ---------------------------------------------------------------------------
extern "C" void kernel_launch(void* const* d_in, const int* in_sizes, int n_in,
                              void* d_out, int out_size, void* d_ws, size_t ws_size,
                              hipStream_t stream) {
  (void)in_sizes; (void)n_in; (void)out_size; (void)ws_size;
  const float* x   = (const float*)d_in[0];
  const int*   am  = (const int*)d_in[1];
  const int*   pos = (const int*)d_in[2];
  const float* wq  = (const float*)d_in[3];
  const float* bq  = (const float*)d_in[4];
  const float* wk  = (const float*)d_in[5];
  const float* bk  = (const float*)d_in[6];
  const float* wv  = (const float*)d_in[7];
  const float* bv  = (const float*)d_in[8];
  const float* wo  = (const float*)d_in[9];
  const float* bo  = (const float*)d_in[10];
  float* out = (float*)d_out;

  char* ws = (char*)d_ws;
  const size_t headBytes = (size_t)Bc * Hc * Sc * HDc * sizeof(__bf16);  // 16 MB
  __bf16* Qw = (__bf16*)(ws);
  __bf16* Kw = (__bf16*)(ws + headBytes);
  __bf16* Vw = (__bf16*)(ws + 2 * headBytes);
  __bf16* Ab = (__bf16*)(ws + 3 * headBytes);

  dim3 g1(Mc / 128, Ec / 128, 3);
  qkv_rope_kernel<<<g1, 256, 0, stream>>>(x, wq, bq, wk, bk, wv, bv, pos, Qw, Kw, Vw);

  dim3 g2(Sc / 128, Hc, Bc);
  attn_kernel<<<g2, 256, 0, stream>>>(Qw, Kw, Vw, am, Ab);

  dim3 g3(Mc / 128, Ec / 128, 1);
  oproj_kernel<<<g3, 256, 0, stream>>>(Ab, wo, bo, out);
}
